// BlockCausalSA_43284680409122
// MI455X (gfx1250) — matrix operand order repaired and checked by reading, not compiled, not run on hardware
//
#include <hip/hip_runtime.h>
#include <hip/hip_bf16.h>

// ---------- types ----------
typedef __attribute__((ext_vector_type(16))) __bf16 bf16x16;
typedef __attribute__((ext_vector_type(8)))  float  f32x8;

union Frag { bf16x16 v; unsigned int u[8]; };

__device__ __forceinline__ unsigned short f2bf(float f) {
    unsigned int x = __float_as_uint(f);
    x += 0x7FFFu + ((x >> 16) & 1u);          // round-to-nearest-even
    return (unsigned short)(x >> 16);
}

// A-fragment (16-bit 16x32) dword index for pair p, lane-half h:
//   pairs p<4 -> K=2p+8h, p>=4 -> K=2p+8+8h   => dword = p + ((p>>2)<<2) + (h<<2)
__device__ __forceinline__ int dwA(int p, int h) { return p + ((p >> 2) << 2) + (h << 2); }
// B-fragment (16-bit 32x16) same mapping as A: dword = p + ((p>>2)<<2) + (h<<2)
__device__ __forceinline__ int dwB(int p, int h) { return p + ((p >> 2) << 2) + (h << 2); }

// ---------- conversion kernels ----------
__global__ void f32_to_bf16_k(const float* __restrict__ in, unsigned short* __restrict__ out, size_t n) {
    for (size_t i = (size_t)blockIdx.x * blockDim.x + threadIdx.x; i < n; i += (size_t)gridDim.x * blockDim.x)
        out[i] = f2bf(in[i]);
}

// in: [K][N] f32 row-major; out: [N][K] bf16 row-major
__global__ void cvt_transpose_k(const float* __restrict__ in, unsigned short* __restrict__ out, int K, int N) {
    size_t total = (size_t)K * N;
    for (size_t i = (size_t)blockIdx.x * blockDim.x + threadIdx.x; i < total; i += (size_t)gridDim.x * blockDim.x) {
        int n = (int)(i / K), k = (int)(i % K);
        out[i] = f2bf(in[(size_t)k * N + n]);
    }
}

// ---------- GEMM: C[M,N](f32) = A[M,K](bf16,row) * Bt[N,K](bf16,row) ----------
// one wave -> one 64(M) x 64(N) tile: 4 A-frags x 4 B-frags -> 16 WMMAs / 32-K step.
// Arithmetic intensity 32 flop/byte of register traffic; B slab reuse 4x vs 16x16.
__global__ void __launch_bounds__(256) gemm_bf16_tn(const unsigned short* __restrict__ A,
                                                    const unsigned short* __restrict__ Bt,
                                                    float* __restrict__ C,
                                                    int M, int N, int K) {
    const int lane = threadIdx.x & 31;
    const int gw   = (int)((blockIdx.x * blockDim.x + threadIdx.x) >> 5);
    const int tilesN = N >> 6;
    const int wm = gw / tilesN;
    const int wn = gw % tilesN;
    if (wm * 64 >= M) return;                      // wave-uniform
    const int half = lane >> 4, l16 = lane & 15;
    const int Kd = K >> 1;
    const unsigned int* Adw = (const unsigned int*)A;
    const unsigned int* Bdw = (const unsigned int*)Bt;

    const f32x8 zero = {0.f, 0.f, 0.f, 0.f, 0.f, 0.f, 0.f, 0.f};
    f32x8 acc[4][4];
#pragma unroll
    for (int mt = 0; mt < 4; ++mt)
#pragma unroll
        for (int t = 0; t < 4; ++t) acc[mt][t] = zero;

    const unsigned int* aps[4];
    const unsigned int* bps[4];
#pragma unroll
    for (int mt = 0; mt < 4; ++mt) aps[mt] = Adw + (size_t)(wm * 64 + mt * 16 + l16) * Kd;
#pragma unroll
    for (int t = 0; t < 4; ++t)    bps[t]  = Bdw + (size_t)(wn * 64 + t * 16 + l16) * Kd;

    for (int k0 = 0; k0 < K; k0 += 32) {
        const int kd = k0 >> 1;
        Frag a[4], b[4];
#pragma unroll
        for (int mt = 0; mt < 4; ++mt)
#pragma unroll
            for (int p = 0; p < 8; ++p) a[mt].u[p] = aps[mt][kd + dwA(p, half)];
#pragma unroll
        for (int t = 0; t < 4; ++t)
#pragma unroll
            for (int p = 0; p < 8; ++p) b[t].u[p] = bps[t][kd + dwB(p, half)];
#pragma unroll
        for (int mt = 0; mt < 4; ++mt)
#pragma unroll
            for (int t = 0; t < 4; ++t)
                acc[mt][t] = __builtin_amdgcn_wmma_f32_16x16x32_bf16(false, a[mt].v, false, b[t].v,
                                                                     (short)0, acc[mt][t], false, false);
    }
#pragma unroll
    for (int mt = 0; mt < 4; ++mt)
#pragma unroll
        for (int t = 0; t < 4; ++t)
#pragma unroll
            for (int e = 0; e < 8; ++e)
                C[(size_t)(wm * 64 + mt * 16 + e + half * 8) * N + (wn * 64 + t * 16 + l16)] = acc[mt][t][e];
}

// ---------- fused RMSNorm + w_ln + RoPE, emit bf16 q/k/v ----------
// qkv: [32768][2304] f32 ; one block (256 thr) per token
__global__ void __launch_bounds__(256) rmsrope_k(const float* __restrict__ qkv,
                                                 const float* __restrict__ w_ln,
                                                 unsigned short* __restrict__ qb,
                                                 unsigned short* __restrict__ kb,
                                                 unsigned short* __restrict__ vb) {
    const int tok = blockIdx.x;
    const float* row = qkv + (size_t)tok * 2304;
    __shared__ float s1[256], s2[256];
    float sq = 0.f, sk = 0.f;
    for (int c = threadIdx.x; c < 768; c += 256) {
        float q = row[c], k = row[768 + c];
        sq += q * q; sk += k * k;
    }
    s1[threadIdx.x] = sq; s2[threadIdx.x] = sk;
    for (int off = 128; off > 0; off >>= 1) {
        __syncthreads();
        if ((int)threadIdx.x < off) { s1[threadIdx.x] += s1[threadIdx.x + off];
                                      s2[threadIdx.x] += s2[threadIdx.x + off]; }
    }
    __syncthreads();
    const float rq = rsqrtf(s1[0] * (1.f / 768.f) + 1e-6f);
    const float rk = rsqrtf(s2[0] * (1.f / 768.f) + 1e-6f);
    const float tpos = (float)(tok & 255);

    for (int c = threadIdx.x; c < 768; c += 256) {
        int jj = c & 63;
        int i  = jj & 31;
        float freq = __powf(10000.0f, -(4.0f * (float)i + 1.0f) * (1.0f / 64.0f));
        float ang = tpos * freq;
        float sn, cs;
        __sincosf(ang, &sn, &cs);
        int cp = c ^ 1;                              // rotation pair partner
        float qn = row[c]        * rq * w_ln[c];
        float kn = row[768 + c]  * rk * w_ln[c];
        float qp = row[cp]       * rq * w_ln[cp];
        float kp = row[768 + cp] * rk * w_ln[cp];
        float rqv = (c & 1) ? qp : -qp;              // out[2i]=-x[2i+1], out[2i+1]=x[2i]
        float rkv = (c & 1) ? kp : -kp;
        size_t o = (size_t)tok * 768 + c;
        qb[o] = f2bf(qn * cs + rqv * sn);
        kb[o] = f2bf(kn * cs + rkv * sn);
        vb[o] = f2bf(row[1536 + c]);
    }
}

// ---------- block-causal attention, one block per (b,g,head) ----------
// q/k/v: [32768][768] bf16 (token-major). 128 threads = 4 waves; each wave owns
// query strips {w, w+4, w+8, w+12} of 16 rows.
__global__ void __launch_bounds__(128) attn_k(const unsigned short* __restrict__ qb,
                                              const unsigned short* __restrict__ kb,
                                              const unsigned short* __restrict__ vb,
                                              unsigned short* __restrict__ yb) {
    const int bid = blockIdx.x;
    const int h   = bid % 12;
    const int blk = bid / 12;
    const size_t tokbase = (size_t)blk * 256;
    const int tid = threadIdx.x;
    const int lane = tid & 31, wave = tid >> 5;
    const int half = lane >> 4, l16 = lane & 15;

    __shared__ unsigned short vT[64][256];      // V transposed: [d][key]   (32 KB)
    __shared__ unsigned short P[4][16][256];    // per-wave probabilities   (32 KB)

    for (int idx = tid; idx < 64 * 256; idx += 128) {
        int d = idx >> 8, key = idx & 255;
        vT[d][key] = vb[(tokbase + key) * 768 + h * 64 + d];
    }
    __syncthreads();

    const unsigned int* qdw = (const unsigned int*)qb;   // row stride 384 dwords
    const unsigned int* kdw = (const unsigned int*)kb;
    const f32x8 zero = {0.f, 0.f, 0.f, 0.f, 0.f, 0.f, 0.f, 0.f};

    for (int i = wave; i < 16; i += 4) {                 // query strip (wave-uniform)
        // Q fragments: K-dim 64 = 2 steps of 32
        Frag qf[2];
        const size_t qrow = (tokbase + i * 16 + l16) * 384 + h * 32;
#pragma unroll
        for (int f = 0; f < 2; ++f)
#pragma unroll
            for (int p = 0; p < 8; ++p) qf[f].u[p] = qdw[qrow + f * 16 + dwA(p, half)];

        // scores S = Q K^T * scale, register-resident
        f32x8 sacc[16];
#pragma unroll
        for (int j = 0; j < 16; ++j) {
            if (j <= i) {                                // wave-uniform branch
                f32x8 c = zero;
                const size_t krow = (tokbase + j * 16 + l16) * 384 + h * 32;
#pragma unroll
                for (int f = 0; f < 2; ++f) {
                    Frag kf;
#pragma unroll
                    for (int p = 0; p < 8; ++p) kf.u[p] = kdw[krow + f * 16 + dwB(p, half)];
                    c = __builtin_amdgcn_wmma_f32_16x16x32_bf16(false, qf[f].v, false, kf.v,
                                                                (short)0, c, false, false);
                }
#pragma unroll
                for (int e = 0; e < 8; ++e) {
                    float s = c[e] * 0.125f;             // 1/sqrt(64)
                    if (j == i && l16 > e + half * 8) s = -3.0e38f;  // causal mask (diag tile)
                    c[e] = s;
                }
                sacc[j] = c;
            } else {
                sacc[j] = zero;
            }
        }

        // softmax: row r = e + 8*half lives in element e across a 16-lane group
        float inv[8];
#pragma unroll
        for (int e = 0; e < 8; ++e) {
            float mm = -3.0e38f;
#pragma unroll
            for (int j = 0; j < 16; ++j) if (j <= i) mm = fmaxf(mm, sacc[j][e]);
#pragma unroll
            for (int off = 8; off > 0; off >>= 1) mm = fmaxf(mm, __shfl_xor(mm, off, 32));
            float ss = 0.f;
#pragma unroll
            for (int j = 0; j < 16; ++j) if (j <= i) { float p = __expf(sacc[j][e] - mm); sacc[j][e] = p; ss += p; }
#pragma unroll
            for (int off = 8; off > 0; off >>= 1) ss += __shfl_xor(ss, off, 32);
            inv[e] = 1.f / ss;
        }

        // write P (bf16) into LDS in row-major order for A-fragment reload
#pragma unroll
        for (int j = 0; j < 16; ++j)
#pragma unroll
            for (int e = 0; e < 8; ++e) {
                float p = (j <= i) ? sacc[j][e] * inv[e] : 0.0f;
                P[wave][e + half * 8][j * 16 + l16] = f2bf(p);
            }

        // y = P @ V : 4 output tiles of 16x16, K=256 in 8 steps of 32
        f32x8 yacc[4] = {zero, zero, zero, zero};
        for (int kk = 0; kk < 8; ++kk) {
            Frag pf;
            const unsigned int* prow = (const unsigned int*)&P[wave][l16][0];
            const int kd0 = kk * 16;                     // 32 keys = 16 dwords
#pragma unroll
            for (int p = 0; p < 8; ++p) pf.u[p] = prow[kd0 + dwA(p, half)];
#pragma unroll
            for (int t = 0; t < 4; ++t) {
                Frag vf;
                const unsigned int* vrow = (const unsigned int*)&vT[t * 16 + l16][0];
#pragma unroll
                for (int p = 0; p < 8; ++p) vf.u[p] = vrow[kd0 + dwB(p, half)];
                yacc[t] = __builtin_amdgcn_wmma_f32_16x16x32_bf16(false, pf.v, false, vf.v,
                                                                  (short)0, yacc[t], false, false);
            }
        }

        // store y strip as bf16, token-major [tok][h*64+d]
#pragma unroll
        for (int t = 0; t < 4; ++t)
#pragma unroll
            for (int e = 0; e < 8; ++e)
                yb[(tokbase + i * 16 + e + half * 8) * 768 + h * 64 + t * 16 + l16] = f2bf(yacc[t][e]);
    }
}

// ---------- launch ----------
extern "C" void kernel_launch(void* const* d_in, const int* in_sizes, int n_in,
                              void* d_out, int out_size, void* d_ws, size_t ws_size,
                              hipStream_t stream) {
    const float* x      = (const float*)d_in[0];   // [4,8192,768]
    const float* w_qkv  = (const float*)d_in[1];   // [768,2304]
    const float* w_ln   = (const float*)d_in[2];   // [768]
    const float* w_proj = (const float*)d_in[3];   // [768,768]
    float* out = (float*)d_out;
    char* ws = (char*)d_ws;

    const size_t NTOK = 32768;
    const size_t SZ_QKVF = NTOK * 2304 * sizeof(float);        // 302 MB
    const size_t SZ_TOKC = NTOK * 768 * sizeof(unsigned short);// 50 MB

    float*          qkvf   = (float*)ws;                       // stage-2 output
    unsigned short* ybf    = (unsigned short*)ws;              // reuses qkvf region after stage 3
    size_t off = SZ_QKVF;
    unsigned short* xbf    = (unsigned short*)(ws + off); off += SZ_TOKC;
    unsigned short* wqkvT  = (unsigned short*)(ws + off); off += (size_t)2304 * 768 * 2;
    unsigned short* wprojT = (unsigned short*)(ws + off); off += (size_t)768 * 768 * 2;
    unsigned short* qbf    = (unsigned short*)(ws + off); off += SZ_TOKC;
    unsigned short* kbf    = (unsigned short*)(ws + off); off += SZ_TOKC;
    unsigned short* vbf    = (unsigned short*)(ws + off);

    f32_to_bf16_k<<<4096, 256, 0, stream>>>(x, xbf, NTOK * 768);
    cvt_transpose_k<<<2048, 256, 0, stream>>>(w_qkv, wqkvT, 768, 2304);
    cvt_transpose_k<<<768, 256, 0, stream>>>(w_proj, wprojT, 768, 768);

    // QKV GEMM: (32768/64)*(2304/64) = 18432 waves -> 2304 blocks of 256
    gemm_bf16_tn<<<2304, 256, 0, stream>>>(xbf, wqkvT, qkvf, 32768, 2304, 768);

    rmsrope_k<<<32768, 256, 0, stream>>>(qkvf, w_ln, qbf, kbf, vbf);

    // attention: 4*32*12 = 1536 (batch, block, head) tiles
    attn_k<<<1536, 128, 0, stream>>>(qbf, kbf, vbf, ybf);

    // proj GEMM: (32768/64)*(768/64) = 6144 waves -> 768 blocks of 256
    gemm_bf16_tn<<<768, 256, 0, stream>>>(ybf, wprojT, out, 32768, 768, 768);
}
